// DCNv2_19481971655393
// MI455X (gfx1250) — compile-verified, matrix operand order checked by reading, not run on hardware
//
#include <hip/hip_runtime.h>
#include <hip/hip_bf16.h>

typedef __attribute__((ext_vector_type(16))) _Float16 v16h;
typedef __attribute__((ext_vector_type(8)))  _Float16 v8h;
typedef __attribute__((ext_vector_type(8)))  float    v8f;

#define B_    4
#define CIN   64
#define H_    128
#define W_    128
#define COUT  64
#define K2    9
#define HO    128
#define WO    128
#define KDIM  576      // CIN * K2, reordered as k = k2*64 + c (tap-major)
#define SW_STRIDE 584  // 576 + 8 halves -> 1168B, 16B aligned
#define SS_STRIDE 328  // 320 + 8 halves -> 656B, 16B aligned (max chunk = 5 taps)

__global__ __launch_bounds__(256) void dcnv2_fused_kernel(
    const float* __restrict__ x,
    const float* __restrict__ off,
    const float* __restrict__ wgt,
    const float* __restrict__ bias,
    float* __restrict__ out)
{
  __shared__ _Float16 sW[COUT * SW_STRIDE];  // 74752 B
  __shared__ _Float16 sS[WO   * SS_STRIDE];  // 83968 B   (total 155 KB)

  const int tid = threadIdx.x;
  const int b   = blockIdx.x >> 7;     // image
  const int i   = blockIdx.x & 127;    // output row

  // ---- stage weights f32 -> f16 into LDS, permuted: sW[m][k2*64 + c] ----
  for (int t = tid; t < COUT * KDIM; t += 256) {
    int m  = t / KDIM;
    int r  = t - m * KDIM;
    int c  = r / K2;
    int k2 = r - c * K2;
    sW[m * SW_STRIDE + k2 * 64 + c] = (_Float16)wgt[t];
  }

  const int lane = tid & 31;           // wave32
  const int wv   = tid >> 5;           // 8 waves per block
  const int pt   = wv * 16;            // this wave's 16-pixel column tile
  const int ko   = (lane >> 4) * 8;    // WMMA 16-bit operand K sub-offset
  const int nl   = lane & 15;

  v8f acc[4] = {};                     // 4 M-tiles of 16x16 f32

  const int p    = tid & 127;          // pixel this thread samples
  const int half = tid >> 7;           // which 32-channel half this thread fills

  // ---- hoist all 18 offset loads (issued together, deep in flight) ----
  float dyv[K2], dxv[K2];
  #pragma unroll
  for (int k2 = 0; k2 < K2; ++k2) {
    dyv[k2] = off[(((b * 18) + 2 * k2    ) * HO + i) * WO + p];
    dxv[k2] = off[(((b * 18) + 2 * k2 + 1) * HO + i) * WO + p];
  }

  const float* xb = x + (size_t)b * (CIN * H_ * W_);

  for (int cc = 0; cc < 2; ++cc) {
    const int t0 = cc ? 4 : 0;         // first tap of chunk
    const int nt = cc ? 5 : 4;         // taps in chunk (K = 256 / 320)
    __syncthreads();                   // protect sS reuse / weight staging

    // ---- bilinear sampling: sS[p][(k2-t0)*64 + c], c contiguous ----
    for (int kt = 0; kt < nt; ++kt) {
      const int k2 = t0 + kt;
      float py = (float)(i - 1 + (k2 / 3)) + dyv[k2];
      float px = (float)(p - 1 + (k2 % 3)) + dxv[k2];
      float y0f = floorf(py), x0f = floorf(px);
      float ly = py - y0f,    lx = px - x0f;
      int y0 = (int)y0f, x0 = (int)x0f;
      int y1 = y0 + 1,   x1 = x0 + 1;
      bool vy0 = (y0 >= 0) & (y0 < H_), vy1 = (y1 >= 0) & (y1 < H_);
      bool vx0 = (x0 >= 0) & (x0 < W_), vx1 = (x1 >= 0) & (x1 < W_);
      int yc0 = min(max(y0, 0), H_ - 1), yc1 = min(max(y1, 0), H_ - 1);
      int xc0 = min(max(x0, 0), W_ - 1), xc1 = min(max(x1, 0), W_ - 1);
      float m00 = (vy0 & vx0) ? (1.f - ly) * (1.f - lx) : 0.f;
      float m01 = (vy0 & vx1) ? (1.f - ly) * lx         : 0.f;
      float m10 = (vy1 & vx0) ? ly * (1.f - lx)         : 0.f;
      float m11 = (vy1 & vx1) ? ly * lx                 : 0.f;
      int o00 = yc0 * W_ + xc0, o01 = yc0 * W_ + xc1;
      int o10 = yc1 * W_ + xc0, o11 = yc1 * W_ + xc1;

      // 4 groups of 8 channels: batch 32 gathers in flight, then compute,
      // then one ds_store_b128 per group.
      #pragma unroll
      for (int g = 0; g < 4; ++g) {
        const int cbase = half * 32 + g * 8;
        float v00[8], v01[8], v10[8], v11[8];
        #pragma unroll
        for (int c = 0; c < 8; ++c) {
          const float* xc = xb + (size_t)(cbase + c) * (H_ * W_);
          v00[c] = xc[o00];
          v01[c] = xc[o01];
          v10[c] = xc[o10];
          v11[c] = xc[o11];
        }
        v8h res;
        #pragma unroll
        for (int c = 0; c < 8; ++c) {
          float s = m00 * v00[c] + m01 * v01[c] + m10 * v10[c] + m11 * v11[c];
          res[c] = (_Float16)s;
        }
        *(v8h*)&sS[p * SS_STRIDE + kt * 64 + cbase] = res;
      }
    }
    __syncthreads();

    // ---- WMMA GEMM over this K chunk ----
    const int nks = cc ? 10 : 8;       // K/32 steps this chunk
    for (int ks = 0; ks < nks; ++ks) {
      const int kb = ks * 32 + ko;     // chunk-local K base for this lane
      const int sbase = (pt + nl) * SS_STRIDE + kb;
      v8h blo = *(const v8h*)&sS[sbase];
      v8h bhi = *(const v8h*)&sS[sbase + 16];
      v16h bf = __builtin_shufflevector(blo, bhi,
                  0,1,2,3,4,5,6,7,8,9,10,11,12,13,14,15);
      const int kw = t0 * 64 + kb;     // global (tap-major) weight K base
      #pragma unroll
      for (int mt = 0; mt < 4; ++mt) {
        const int abase = (mt * 16 + nl) * SW_STRIDE + kw;
        v8h alo = *(const v8h*)&sW[abase];
        v8h ahi = *(const v8h*)&sW[abase + 16];
        v16h af = __builtin_shufflevector(alo, ahi,
                    0,1,2,3,4,5,6,7,8,9,10,11,12,13,14,15);
        acc[mt] = __builtin_amdgcn_wmma_f32_16x16x32_f16(
            /*neg_a=*/false, af, /*neg_b=*/false, bf,
            /*c_mod=*/(short)0, acc[mt],
            /*reuse_a=*/false, /*reuse_b=*/false);
      }
    }
  }

  // ---- epilogue: C/D layout -> out[b][cout][i][j], add bias ----
  const int mrow = (lane >> 4) * 8;
  const int jcol = pt + (lane & 15);
  #pragma unroll
  for (int mt = 0; mt < 4; ++mt) {
    #pragma unroll
    for (int r = 0; r < 8; ++r) {
      int co = mt * 16 + mrow + r;
      out[(((size_t)b * COUT + co) * HO + i) * WO + jcol] = acc[mt][r] + bias[co];
    }
  }
}

extern "C" void kernel_launch(void* const* d_in, const int* in_sizes, int n_in,
                              void* d_out, int out_size, void* d_ws, size_t ws_size,
                              hipStream_t stream) {
  const float* x    = (const float*)d_in[0];
  const float* off  = (const float*)d_in[1];
  // d_in[2] = mask: unused (reference's _deform_conv ignores it)
  const float* wgt  = (const float*)d_in[3];
  const float* bias = (const float*)d_in[4];
  float* out = (float*)d_out;

  dim3 grid(B_ * HO);   // 512 blocks: one (image, output-row) each
  dim3 block(256);      // 8 wave32 waves
  dcnv2_fused_kernel<<<grid, block, 0, stream>>>(x, off, wgt, bias, out);
}
